// pMoE_36799279792635
// MI455X (gfx1250) — matrix operand order repaired and checked by reading, not compiled, not run on hardware
//
#include <hip/hip_runtime.h>
#include <hip/hip_bf16.h>
#include <math.h>

// ---------------- problem constants ----------------
#define NB     8192      // tokens
#define ND     1024      // model dim
#define NDFF   1024      // ffn dim
#define NE     32        // experts
#define NTOPK  2
#define NP     (NB * NTOPK)   // dispatched rows = 16384

// GEMM tiling
#define BM 128
#define BN 128
#define BK 32
#define APITCH 40        // LDS pitch (bf16): TDM pads 64B rows with 16B -> 80B pitch
#define BPITCH 40
#define LDSBUF (BM * APITCH * 2)   // bytes per buffer (= 10240), same for A and B
#define NSTAGE (ND / BK)           // 32
#define MAXT 160         // max grouped-GEMM tasks: NP/BM + NE

// workspace metadata layout (int indices into (int*)d_ws)
#define IDX_COUNTS 0       // 32 ints
#define IDX_OFFS   64      // 33 ints
#define IDX_NTASKS 128     // 1 int
#define IDX_TASKE  256     // up to MAXT ints
#define IDX_TASKM  512     // up to MAXT ints
#define IDX_SLOTOFPOS 1024 // NP ints   (byte 4096)
#define IDX_EXPOFSLOT 32768// NP ints   (byte 131072)
#define WS_XD_BYTES   262144                            // bf16 [NP][ND]
#define WS_H_BYTES    (WS_XD_BYTES + (size_t)NP*ND*2)   // bf16 [NP][NDFF]
#define WS_W1T_BYTES  (WS_H_BYTES + (size_t)NP*NDFF*2)  // bf16 [NE][NDFF][ND]  (n-major, k contiguous)
#define WS_W2T_BYTES  (WS_W1T_BYTES + (size_t)NE*ND*NDFF*2) // bf16 [NE][ND][NDFF]

typedef __bf16 bf16_t;
typedef bf16_t v16bf __attribute__((ext_vector_type(16)));
typedef float  v8f   __attribute__((ext_vector_type(8)));
typedef unsigned int v4u __attribute__((ext_vector_type(4)));
typedef int v4i __attribute__((ext_vector_type(4)));
typedef int v8i __attribute__((ext_vector_type(8)));

union FragBF16 {
    uint4 q[2];
    v16bf v;
};

__device__ __forceinline__ unsigned short f2bf(float f) {
    unsigned int u = __float_as_uint(f);
    unsigned int r = u + 0x7FFFu + ((u >> 16) & 1u);   // round-to-nearest-even
    return (unsigned short)(r >> 16);
}

__device__ __forceinline__ float gelu_exact(float x) {
    return 0.5f * x * (1.0f + erff(x * 0.70710678118654752f));
}

// ---- CDNA5 Tensor Data Mover: one instruction moves a 2-D tile global->LDS.
// Tile: tile_dim0=32 bf16 (64B rows, k-contiguous) x tile_dim1=128 rows,
// row stride = 1024 elems; LDS padded 16B per 64B row -> 80B pitch (APITCH=40).
// Rows >= nrows are hardware zero-filled (ragged expert boundary).
__device__ __forceinline__ void tdm_load_tile(unsigned lds_off, const void* gptr,
                                              unsigned nrows) {
    unsigned long long ga = (unsigned long long)(uintptr_t)gptr;
    v4u g0;
    g0[0] = 1u;                                             // count=1, user descriptor
    g0[1] = lds_off;                                        // lds_addr (bytes)
    g0[2] = (unsigned)(ga & 0xFFFFFFFFu);                   // global_addr[31:0]
    g0[3] = (unsigned)((ga >> 32) & 0x01FFFFFFu) | 0x80000000u; // addr[56:32] | type=2
    v8i g1;
    g1[0] = (int)0x06D10000u;       // data_size=2B | pad_enable | pad_interval=16dw | pad_amount=4dw
    g1[1] = (int)(1024u << 16);     // tensor_dim0 = 1024 (elems), low 16 bits
    g1[2] = (int)(nrows << 16);     // tensor_dim0 hi | tensor_dim1 lo = nrows
    g1[3] = (int)(32u << 16);       // tensor_dim1 hi | tile_dim0 = 32
    g1[4] = (int)128;               // tile_dim1 = 128, tile_dim2 = 0
    g1[5] = (int)1024;              // tensor_dim0_stride = 1024 elems
    g1[6] = 0;
    g1[7] = 0;
    v4i z4 = {0, 0, 0, 0};          // 2-D tensor: groups 2/3 unused
#if defined(__clang_major__) && (__clang_major__ >= 23)
    v8i z8 = {0, 0, 0, 0, 0, 0, 0, 0};
    __builtin_amdgcn_tensor_load_to_lds(g0, g1, z4, z4, z8, 0);
#else
    __builtin_amdgcn_tensor_load_to_lds(g0, g1, z4, z4, 0);
#endif
}
__device__ __forceinline__ void tdm_wait0() {
    __builtin_amdgcn_s_wait_tensorcnt((short)0);
}

// ---------------- kernel 0: zero output + metadata ----------------
__global__ __launch_bounds__(256) void pmoe_zero_kernel(float* __restrict__ out, int n4,
                                                        int* __restrict__ meta) {
    int i = blockIdx.x * blockDim.x + threadIdx.x;
    if (i < n4) {
        float4 z = {0.f, 0.f, 0.f, 0.f};
        reinterpret_cast<float4*>(out)[i] = z;
    }
    if (blockIdx.x == 0 && threadIdx.x < 1024) {
        meta[threadIdx.x] = 0;   // counts/offsets/ntasks region
    }
}

// ---------------- weight convert + transpose: W[e][k][n] f32 -> WT[e][n][k] bf16 ----------------
__global__ __launch_bounds__(256) void pmoe_wtrans_kernel(const float* __restrict__ W,
                                                          unsigned short* __restrict__ WT) {
    __shared__ alignas(16) float T[64][68];
    const int e  = blockIdx.y;
    const int kt = (blockIdx.x & 15) * 64;
    const int nt = (blockIdx.x >> 4) * 64;
    const int tid = threadIdx.x;

    const float* src = W + (size_t)e * ND * NDFF + (size_t)kt * NDFF + nt;
#pragma unroll
    for (int i = 0; i < 4; ++i) {            // 64x64 f32 tile, coalesced along n
        int g = i * 256 + tid;
        int r = g >> 4;
        int c = (g & 15) * 4;
        float4 v = *reinterpret_cast<const float4*>(src + (size_t)r * NDFF + c);
        T[r][c + 0] = v.x; T[r][c + 1] = v.y; T[r][c + 2] = v.z; T[r][c + 3] = v.w;
    }
    __syncthreads();

    unsigned short* dst = WT + (size_t)e * ND * NDFF + (size_t)nt * ND + kt;
#pragma unroll
    for (int i = 0; i < 2; ++i) {            // write bf16 granules of 8, contiguous along k
        int g  = i * 256 + tid;
        int n  = g >> 3;
        int k8 = (g & 7) * 8;
        unsigned int q0 = (unsigned)f2bf(T[k8 + 0][n]) | ((unsigned)f2bf(T[k8 + 1][n]) << 16);
        unsigned int q1 = (unsigned)f2bf(T[k8 + 2][n]) | ((unsigned)f2bf(T[k8 + 3][n]) << 16);
        unsigned int q2 = (unsigned)f2bf(T[k8 + 4][n]) | ((unsigned)f2bf(T[k8 + 5][n]) << 16);
        unsigned int q3 = (unsigned)f2bf(T[k8 + 6][n]) | ((unsigned)f2bf(T[k8 + 7][n]) << 16);
        uint4 o = {q0, q1, q2, q3};
        *reinterpret_cast<uint4*>(dst + (size_t)n * ND + k8) = o;
    }
}

// ---------------- kernel 1: gating (one wave per token) ----------------
__global__ __launch_bounds__(256) void pmoe_gate_kernel(const float* __restrict__ X,
                                                        const float* __restrict__ Wg,
                                                        const float* __restrict__ bg,
                                                        int* __restrict__ outIdx,
                                                        int* __restrict__ counts,
                                                        int* __restrict__ expert_of_slot) {
    const int lane = threadIdx.x;                // 0..31  (wave32)
    const int tok  = blockIdx.x * blockDim.y + threadIdx.y;
    const float* x = X + (size_t)tok * ND;

    float acc = bg[lane];
    for (int d0 = 0; d0 < ND; d0 += 32) {
        float xv = x[d0 + lane];
#pragma unroll
        for (int j = 0; j < 32; ++j) {
            float xj = __shfl(xv, j, 32);
            acc = fmaf(xj, Wg[(size_t)(d0 + j) * NE + lane], acc);
        }
    }

    // top-1 (ties -> lowest index, matching jax.lax.top_k)
    float v1 = acc; int i1 = lane;
#pragma unroll
    for (int o = 16; o >= 1; o >>= 1) {
        float ov = __shfl_xor(v1, o, 32);
        int   oi = __shfl_xor(i1, o, 32);
        if (ov > v1 || (ov == v1 && oi < i1)) { v1 = ov; i1 = oi; }
    }
    // top-2
    float mv = (lane == i1) ? -3.4e38f : acc;
    float v2 = mv; int i2 = lane;
#pragma unroll
    for (int o = 16; o >= 1; o >>= 1) {
        float ov = __shfl_xor(v2, o, 32);
        int   oi = __shfl_xor(i2, o, 32);
        if (ov > v2 || (ov == v2 && oi < i2)) { v2 = ov; i2 = oi; }
    }

    if (lane == 0) {
        outIdx[tok * 2 + 0] = i1;
        outIdx[tok * 2 + 1] = i2;
        expert_of_slot[tok * 2 + 0] = i1;
        expert_of_slot[tok * 2 + 1] = i2;
        atomicAdd(&counts[i1], 1);
        atomicAdd(&counts[i2], 1);
    }
}

// ---------------- kernel 2: offsets + tile task list (serial, tiny) ----------------
__global__ void pmoe_scan_kernel(int* __restrict__ meta) {
    if (threadIdx.x != 0) return;
    int off = 0, nt = 0;
    for (int e = 0; e < NE; ++e) {
        meta[IDX_OFFS + e] = off;
        int c = meta[IDX_COUNTS + e];
        for (int m = 0; m < c; m += BM) {
            meta[IDX_TASKE + nt] = e;
            meta[IDX_TASKM + nt] = m;
            ++nt;
        }
        off += c;
    }
    meta[IDX_OFFS + NE] = off;     // == NP
    meta[IDX_NTASKS]    = nt;
}

// ---------------- kernel 3: stable per-expert rank -> slot_of_pos ----------------
__global__ __launch_bounds__(32) void pmoe_rank_kernel(const int* __restrict__ eos,
                                                       const int* __restrict__ meta,
                                                       int* __restrict__ slot_of_pos) {
    const int e    = blockIdx.x;
    const int lane = threadIdx.x;
    int base = meta[IDX_OFFS + e];
    for (int s0 = 0; s0 < NP; s0 += 32) {
        int  s = s0 + lane;
        bool f = (eos[s] == e);
        unsigned int mask = (unsigned int)__ballot(f);
        if (f) {
            int r = __popc(mask & ((1u << lane) - 1u));
            slot_of_pos[base + r] = s;
        }
        base += __popc(mask);
    }
}

// ---------------- kernel 4: gather dispatched rows, f32 -> bf16 ----------------
__global__ __launch_bounds__(128) void pmoe_gather_kernel(const float* __restrict__ X,
                                                          const int* __restrict__ slot_of_pos,
                                                          unsigned short* __restrict__ Xd) {
    const int p = blockIdx.x;
    const int t = threadIdx.x;                 // 128 threads * 8 f32 = 1024
    const int tok = slot_of_pos[p] >> 1;       // slot/TOPK
    const float4* src = reinterpret_cast<const float4*>(X + (size_t)tok * ND) + t * 2;
    float4 a = src[0];
    float4 b = src[1];
    unsigned int p0 = (unsigned int)f2bf(a.x) | ((unsigned int)f2bf(a.y) << 16);
    unsigned int p1 = (unsigned int)f2bf(a.z) | ((unsigned int)f2bf(a.w) << 16);
    unsigned int p2 = (unsigned int)f2bf(b.x) | ((unsigned int)f2bf(b.y) << 16);
    unsigned int p3 = (unsigned int)f2bf(b.z) | ((unsigned int)f2bf(b.w) << 16);
    uint4 o = {p0, p1, p2, p3};
    *reinterpret_cast<uint4*>(Xd + (size_t)p * ND + t * 8) = o;
}

// ---------------- grouped GEMM: C = A(bf16,k-major) x WT_e(bf16,k-major)^T ----------------
// Operand tiles staged by the Tensor Data Mover (wave 0 issues 2 TDM ops per stage,
// TENSORcnt-tracked, overlapped with the WMMA block of the previous stage).
// PHASE 1: H = bf16( gelu(C + b1[e]) )
// PHASE 2: atomicAdd(out[row>>1], C + b2[e])
template <int PHASE>
__global__ __launch_bounds__(256) void pmoe_gemm_kernel(const unsigned short* __restrict__ Abf,
                                                        const unsigned short* __restrict__ WT,
                                                        const float* __restrict__ bias,
                                                        unsigned short* __restrict__ Hout,
                                                        float* __restrict__ Yout,
                                                        const int* __restrict__ meta) {
    const int task = blockIdx.x;
    if (task >= meta[IDX_NTASKS]) return;

    const int e    = meta[IDX_TASKE + task];
    const int mrel = meta[IDX_TASKM + task];
    const int offe = meta[IDX_OFFS + e];
    const int cnte = meta[IDX_COUNTS + e];
    const int rowcnt  = min(BM, cnte - mrel);
    const int rowbase = offe + mrel;
    const int n0 = blockIdx.y * BN;

    const int tid  = threadIdx.x;
    const int lane = tid & 31;
    const int wid  = tid >> 5;        // 8 waves
    const int wr   = wid >> 2;        // 0..1 : wave M block (64 rows)
    const int wc   = wid & 3;         // 0..3 : wave N block (32 cols)

    __shared__ alignas(16) unsigned short As[2][BM * APITCH];
    __shared__ alignas(16) unsigned short Bs[2][BN * BPITCH];

    const unsigned asBase = (unsigned)(uintptr_t)(&As[0][0]);
    const unsigned bsBase = (unsigned)(uintptr_t)(&Bs[0][0]);
    const unsigned short* gA = Abf + (size_t)rowbase * ND;                 // tile row 0, k advances
    const unsigned short* gB = WT + (size_t)e * ND * NDFF + (size_t)n0 * ND;

    // named accumulators: keep WMMA D=C tied in place across the pipeline
    v8f c00 = {}, c01 = {}, c10 = {}, c11 = {}, c20 = {}, c21 = {}, c30 = {}, c31 = {};

    const int klo = (lane >> 4) * 8;    // A frag: lane<16 K{0..7,16..23}; lane>=16 K{8..15,24..31}
    const int kb2 = (lane >> 4) * 8;    // B frag: lane<16 K{0..7,16..23}; lane>=16 K{8..15,24..31}
    const int ar0 = (wr * 64 + (lane & 15)) * APITCH + klo;
    const int bc0 = (wc * 32 + (lane & 15)) * BPITCH + kb2;

    auto compute = [&](int buf) {
        const unsigned short* as = &As[buf][0];
        const unsigned short* bs = &Bs[buf][0];
        FragBF16 a0, a1, a2, a3, b0, b1;
        a0.q[0] = *reinterpret_cast<const uint4*>(as + ar0);
        a0.q[1] = *reinterpret_cast<const uint4*>(as + ar0 + 16);
        a1.q[0] = *reinterpret_cast<const uint4*>(as + ar0 + 16 * APITCH);
        a1.q[1] = *reinterpret_cast<const uint4*>(as + ar0 + 16 * APITCH + 16);
        a2.q[0] = *reinterpret_cast<const uint4*>(as + ar0 + 32 * APITCH);
        a2.q[1] = *reinterpret_cast<const uint4*>(as + ar0 + 32 * APITCH + 16);
        a3.q[0] = *reinterpret_cast<const uint4*>(as + ar0 + 48 * APITCH);
        a3.q[1] = *reinterpret_cast<const uint4*>(as + ar0 + 48 * APITCH + 16);
        b0.q[0] = *reinterpret_cast<const uint4*>(bs + bc0);
        b0.q[1] = *reinterpret_cast<const uint4*>(bs + bc0 + 16);
        b1.q[0] = *reinterpret_cast<const uint4*>(bs + bc0 + 16 * BPITCH);
        b1.q[1] = *reinterpret_cast<const uint4*>(bs + bc0 + 16 * BPITCH + 16);
        c00 = __builtin_amdgcn_wmma_f32_16x16x32_bf16(false, a0.v, false, b0.v, (short)0, c00, false, false);
        c01 = __builtin_amdgcn_wmma_f32_16x16x32_bf16(false, a0.v, false, b1.v, (short)0, c01, false, false);
        c10 = __builtin_amdgcn_wmma_f32_16x16x32_bf16(false, a1.v, false, b0.v, (short)0, c10, false, false);
        c11 = __builtin_amdgcn_wmma_f32_16x16x32_bf16(false, a1.v, false, b1.v, (short)0, c11, false, false);
        c20 = __builtin_amdgcn_wmma_f32_16x16x32_bf16(false, a2.v, false, b0.v, (short)0, c20, false, false);
        c21 = __builtin_amdgcn_wmma_f32_16x16x32_bf16(false, a2.v, false, b1.v, (short)0, c21, false, false);
        c30 = __builtin_amdgcn_wmma_f32_16x16x32_bf16(false, a3.v, false, b0.v, (short)0, c30, false, false);
        c31 = __builtin_amdgcn_wmma_f32_16x16x32_bf16(false, a3.v, false, b1.v, (short)0, c31, false, false);
    };
    auto issue_stage = [&](int s, int buf) {    // wave 0 only: 2 TDM tile moves
        const unsigned bo = (unsigned)buf * LDSBUF;
        const int      kb = s * BK;
        tdm_load_tile(asBase + bo, gA + kb, (unsigned)rowcnt);   // ragged rows -> zero-filled
        tdm_load_tile(bsBase + bo, gB + kb, (unsigned)BN);
    };

    if (wid == 0) {
        issue_stage(0, 0);
        tdm_wait0();
    }
    __syncthreads();

    for (int s = 0; s < NSTAGE; s += 2) {
        if (wid == 0) issue_stage(s + 1, 1);    // DMA s+1 overlaps WMMAs of s
        compute(0);
        if (wid == 0) tdm_wait0();
        __syncthreads();

        if (wid == 0 && s + 2 < NSTAGE) issue_stage(s + 2, 0);
        compute(1);
        if (wid == 0 && s + 2 < NSTAGE) tdm_wait0();
        __syncthreads();
    }

    // ---- epilogue
    v8f accs[4][2] = {{c00, c01}, {c10, c11}, {c20, c21}, {c30, c31}};
    const int half  = lane >> 4;
    const int nlane = lane & 15;
#pragma unroll
    for (int j = 0; j < 2; ++j) {
        const int ncol = n0 + wc * 32 + j * 16 + nlane;
        const float bv = bias[(size_t)e * NDFF + ncol];
#pragma unroll
        for (int i = 0; i < 4; ++i) {
#pragma unroll
            for (int r = 0; r < 8; ++r) {
                int m = wr * 64 + i * 16 + half * 8 + r;
                if (m < rowcnt) {
                    float v = accs[i][j][r] + bv;
                    if (PHASE == 1) {
                        v = gelu_exact(v);
                        Hout[(size_t)(rowbase + m) * NDFF + ncol] = f2bf(v);
                    } else {
                        // pair-sum: dispatch rows 2i,2i+1 -> out row i (2 commutative adds, deterministic)
                        atomicAdd(&Yout[(size_t)((rowbase + m) >> 1) * ND + ncol], v);
                    }
                }
            }
        }
    }
}

// ---------------- launcher ----------------
extern "C" void kernel_launch(void* const* d_in, const int* in_sizes, int n_in,
                              void* d_out, int out_size, void* d_ws, size_t ws_size,
                              hipStream_t stream) {
    const float* moe_inp = (const float*)d_in[0];
    const float* wg      = (const float*)d_in[1];
    const float* bg      = (const float*)d_in[2];
    const float* w1      = (const float*)d_in[3];
    const float* b1      = (const float*)d_in[4];
    const float* w2      = (const float*)d_in[5];
    const float* b2      = (const float*)d_in[6];

    float* out    = (float*)d_out;
    int*   outIdx = (int*)(out + (size_t)NB * ND);   // gate_top_k_idx region

    int* meta        = (int*)d_ws;
    int* slot_of_pos = meta + IDX_SLOTOFPOS;
    int* eos         = meta + IDX_EXPOFSLOT;
    unsigned short* Xd  = (unsigned short*)((char*)d_ws + WS_XD_BYTES);
    unsigned short* H   = (unsigned short*)((char*)d_ws + WS_H_BYTES);
    unsigned short* W1T = (unsigned short*)((char*)d_ws + WS_W1T_BYTES);
    unsigned short* W2T = (unsigned short*)((char*)d_ws + WS_W2T_BYTES);

    // 0) zero output accumulator + metadata
    pmoe_zero_kernel<<<(NB * ND / 4 + 255) / 256, 256, 0, stream>>>(out, NB * ND / 4, meta);
    // 0b) weights -> bf16, transposed to [e][n][k] so GEMM B-tiles are k-contiguous
    pmoe_wtrans_kernel<<<dim3(256, NE), 256, 0, stream>>>(w1, W1T);
    pmoe_wtrans_kernel<<<dim3(256, NE), 256, 0, stream>>>(w2, W2T);
    // 1) gate: logits, top-2, counts
    pmoe_gate_kernel<<<NB / 8, dim3(32, 8), 0, stream>>>(moe_inp, wg, bg, outIdx, meta + IDX_COUNTS, eos);
    // 2) offsets + tile tasks
    pmoe_scan_kernel<<<1, 32, 0, stream>>>(meta);
    // 3) stable expert-sorted permutation
    pmoe_rank_kernel<<<NE, 32, 0, stream>>>(eos, meta, slot_of_pos);
    // 4) gather + convert activations to bf16
    pmoe_gather_kernel<<<NP, 128, 0, stream>>>(moe_inp, slot_of_pos, Xd);
    // 5) H = gelu(Xd @ w1[e] + b1[e])
    pmoe_gemm_kernel<1><<<dim3(MAXT, NDFF / BN), 256, 0, stream>>>(Xd, W1T, b1, H, nullptr, meta);
    // 6) out[p>>1] += H @ w2[e] + b2[e]
    pmoe_gemm_kernel<2><<<dim3(MAXT, ND / BN), 256, 0, stream>>>(H, W2T, b2, nullptr, out, meta);
}
